// GroupedQueryAttention_86689619903473
// MI455X (gfx1250) — matrix operand order repaired and checked by reading, not compiled, not run on hardware
//
#include <hip/hip_runtime.h>
#include <hip/hip_bf16.h>

// ---------------------------------------------------------------------------
// Grouped-Query Attention for MI455X (gfx1250), bf16 WMMA datapath + TDM.
// B=4, T=2048, D=2048, 16 heads / 4 KV heads, head_dim=128, RoPE, causal.
// ---------------------------------------------------------------------------

static constexpr int Bsz = 4;
static constexpr int Tsz = 2048;
static constexpr int Dsz = 2048;
static constexpr int NH  = 16;
static constexpr int NKV = 4;
static constexpr int HDm = 128;

typedef __attribute__((ext_vector_type(16))) __bf16 bf16x16;
typedef __attribute__((ext_vector_type(8)))  __bf16 bf16x8;
typedef __attribute__((ext_vector_type(8)))  float  f32x8;
typedef __attribute__((ext_vector_type(4)))  unsigned int u32x4;
typedef __attribute__((ext_vector_type(8)))  int i32x8;
typedef __attribute__((ext_vector_type(4)))  int i32x4;

union ABf16 {            // one WMMA 16-bit operand = 8 VGPRs = 16 bf16 / lane
  bf16x16 v;
  bf16x8  h[2];
};

__device__ __forceinline__ f32x8 wmma_bf16(bf16x16 a, bf16x16 b, f32x8 c) {
  // (neg_a, A, neg_b, B, c_mod, C, reuse_a, reuse_b)
  return __builtin_amdgcn_wmma_f32_16x16x32_bf16(false, a, false, b,
                                                 (short)0, c, false, false);
}

__device__ __forceinline__ f32x8 fzero8() {
  f32x8 z = {0.f, 0.f, 0.f, 0.f, 0.f, 0.f, 0.f, 0.f};
  return z;
}

#if defined(__gfx1250__) && __has_builtin(__builtin_amdgcn_tensor_load_to_lds)
#define USE_TDM 1
#else
#define USE_TDM 0
#endif

// ---------------------------------------------------------------------------
// f32 -> bf16 elementwise convert
// ---------------------------------------------------------------------------
__global__ void cvt_f32_to_bf16(const float* __restrict__ in,
                                __bf16* __restrict__ out, int n) {
  int i = blockIdx.x * blockDim.x + threadIdx.x;
  if (i < n) out[i] = (__bf16)in[i];
}

// ---------------------------------------------------------------------------
// GEMM: C[M,N] = A[M,K] * W[N,K]^T   (A, W bf16; C f32)
// Block = 128 threads = 4 waves. Wave tile: 64(M) x 32(N), K-step 32.
// The block's shared 32xK W slice is staged through LDS by the Tensor Data
// Mover in 32x256 tiles (D# per ISA 8.3/8.4), tracked with TENSORcnt.
// A-operand lane layout (16-bit A 16x32, same k order as B):
//   lane = m (0..15), half h = lane>>4; elems e -> k = kb+16h+e
//   (one contiguous 32B chunk per lane)    => two 16B loads.
// B-operand (16-bit B 32x16): lane = n, half h; elems e -> k = kb+16h+e
//   => one contiguous 32B chunk (two 16B loads, here from LDS).
// C layout: lane = n (mod 16), vgpr r -> m = r + 8*(lane>>4).
// ---------------------------------------------------------------------------
__global__ __launch_bounds__(128) void gemm_bf16_nt(
    const __bf16* __restrict__ A, const __bf16* __restrict__ W,
    float* __restrict__ C, int M, int N, int K)
{
  __shared__ __align__(128) __bf16 bstage[32 * 256];   // 16 KB W tile
  const int lane = threadIdx.x & 31;
  const int wave = threadIdx.x >> 5;
  const int c  = lane & 15;
  const int hh = lane >> 4;
  const int n0 = blockIdx.x * 32;
  const int m0 = blockIdx.y * 256 + wave * 64;

  f32x8 acc[4][2];
#pragma unroll
  for (int mt = 0; mt < 4; ++mt)
#pragma unroll
    for (int nt = 0; nt < 2; ++nt) acc[mt][nt] = fzero8();

  const __bf16* arow[4];
#pragma unroll
  for (int mt = 0; mt < 4; ++mt)
    arow[mt] = A + (size_t)(m0 + mt * 16 + c) * K;

  for (int kk0 = 0; kk0 < K; kk0 += 256) {
#if USE_TDM
    if (wave == 0) {
      // ---- D# group 0: count=1, lds_addr, global_addr, type=2 ----
      const unsigned long long ga =
          (unsigned long long)(const void*)(W + (size_t)n0 * K + kk0);
      const unsigned ldsoff = (unsigned)(__SIZE_TYPE__)(&bstage[0]);
      u32x4 g0;
      g0[0] = 1u;                                    // count = 1 (valid D#)
      g0[1] = ldsoff;                                // lds_addr (bytes)
      g0[2] = (unsigned)(ga & 0xffffffffu);          // global_addr[95:64]
      g0[3] = (unsigned)((ga >> 32) & 0x01ffffffu)   // global_addr[120:96]
              | 0x80000000u;                         // type = 2 ("image")
      // ---- D# group 1: data_size=2B, dims/strides, 32x256 tile ----
      i32x8 g1;
      g1[0] = 0x00010000;                                  // data_size=1 -> 2B
      g1[1] = (int)(((unsigned)K & 0xffffu) << 16);        // tensor_dim0 lo16
      g1[2] = (int)((((unsigned)K >> 16) & 0xffffu)        // tensor_dim0 hi16
                    | (((unsigned)N & 0xffffu) << 16));    // tensor_dim1 lo16
      g1[3] = (int)((((unsigned)N >> 16) & 0xffffu)        // tensor_dim1 hi16
                    | (256u << 16));                       // tile_dim0 = 256
      g1[4] = 32;                                          // tile_dim1 = 32
      g1[5] = K;                                           // dim0_stride lo32
      g1[6] = 0;
      g1[7] = 0;
      i32x4 g2 = {0, 0, 0, 0};                             // 2D tensor
      i32x4 g3 = {0, 0, 0, 0};
      i32x8 g4 = {0, 0, 0, 0, 0, 0, 0, 0};                 // extra operand
      // amdgpu-toolchain (clang-23) 6-arg form:
      // (u32x4 g0, i32x8 g1, i32x4 g2, i32x4 g3, i32x8, i32 cpol)
      __builtin_amdgcn_tensor_load_to_lds(g0, g1, g2, g3, g4, 0);
      __builtin_amdgcn_s_wait_tensorcnt(0);
    }
#else
    // Fallback: cooperative direct loads (128 lanes x 128B = 16 KB tile)
    for (int e = threadIdx.x; e < 32 * 256 / 8; e += 128) {
      const int rr = e / 32;            // tile row (n local)
      const int ee = (e % 32) * 8;      // k offset (8 bf16)
      *(bf16x8*)(&bstage[rr * 256 + ee]) =
          *(const bf16x8*)(W + (size_t)(n0 + rr) * K + kk0 + ee);
    }
#endif
    __syncthreads();

    for (int ks = 0; ks < 256; ks += 32) {
      const int kk = kk0 + ks;
      if (kk + 32 < K)
        __builtin_prefetch(arow[0] + kk + 32, 0, 3);

      ABf16 b[2];
#pragma unroll
      for (int nt = 0; nt < 2; ++nt) {
        const __bf16* pb = &bstage[(nt * 16 + c) * 256 + ks + 16 * hh];
        b[nt].h[0] = *(const bf16x8*)(pb);
        b[nt].h[1] = *(const bf16x8*)(pb + 8);
      }
      ABf16 a[4];
#pragma unroll
      for (int mt = 0; mt < 4; ++mt) {
        const __bf16* pa = arow[mt] + kk + 16 * hh;
        a[mt].h[0] = *(const bf16x8*)(pa);
        a[mt].h[1] = *(const bf16x8*)(pa + 8);
      }
#pragma unroll
      for (int mt = 0; mt < 4; ++mt)
#pragma unroll
        for (int nt = 0; nt < 2; ++nt)
          acc[mt][nt] = wmma_bf16(a[mt].v, b[nt].v, acc[mt][nt]);
    }
    __syncthreads();
  }

#pragma unroll
  for (int mt = 0; mt < 4; ++mt)
#pragma unroll
    for (int nt = 0; nt < 2; ++nt)
#pragma unroll
      for (int r = 0; r < 8; ++r) {
        const int row = m0 + mt * 16 + r + 8 * hh;
        const int col = n0 + nt * 16 + c;
        C[(size_t)row * N + col] = acc[mt][nt][r];
      }
}

// ---------------------------------------------------------------------------
// RoPE + pack f32 [B*T, nheads*HD] -> bf16 [B, nheads, T, HD]
// pair (i, i+64), inv_freq = theta^(-i/64)
// ---------------------------------------------------------------------------
__global__ void rope_pack(const float* __restrict__ src,
                          __bf16* __restrict__ dst, int nheads) {
  int idx = blockIdx.x * blockDim.x + threadIdx.x;
  int total = Bsz * nheads * Tsz * (HDm / 2);
  if (idx >= total) return;
  const int i = idx & 63;
  const int t = (idx >> 6) & (Tsz - 1);
  const int rest = idx >> 17;
  const int hh = rest % nheads;
  const int b  = rest / nheads;

  const float* row = src + ((size_t)(b * Tsz + t)) * (nheads * HDm) + hh * HDm;
  const float x1 = row[i];
  const float x2 = row[i + 64];
  const float inv = __powf(10000.0f, -(float)i * (1.0f / 64.0f));
  const float ang = (float)t * inv;
  float sn, cs;
  __sincosf(ang, &sn, &cs);
  __bf16* drow = dst + (((size_t)(b * nheads + hh)) * Tsz + t) * HDm;
  drow[i]      = (__bf16)(x1 * cs - x2 * sn);
  drow[i + 64] = (__bf16)(x1 * sn + x2 * cs);
}

// ---------------------------------------------------------------------------
// V pack+transpose: f32 [B*T, NKV*HD] -> bf16 [B, NKV, HD, T]
// (so the P*V B-operand becomes a contiguous 32B load per lane)
// ---------------------------------------------------------------------------
__global__ void v_pack_T(const float* __restrict__ vf, __bf16* __restrict__ vT) {
  int idx = blockIdx.x * blockDim.x + threadIdx.x;
  const int t  = idx & (Tsz - 1);
  const int hd = (idx >> 11) & (HDm - 1);
  const int kv = (idx >> 18) & (NKV - 1);
  const int b  = idx >> 20;
  if (b >= Bsz) return;
  const float val = vf[((size_t)(b * Tsz + t)) * (NKV * HDm) + kv * HDm + hd];
  vT[idx] = (__bf16)val;
}

// ---------------------------------------------------------------------------
// Fused causal flash attention, one wave (32 threads) per (b, head, 16-row
// q-tile). Online softmax over 32-wide KV blocks:
//   S (2x 16x16 f32) via 8 WMMAs, P repacked C-layout -> A-layout through LDS,
//   O += P*V via 8 WMMAs into 16x128 f32 accumulators.
// ---------------------------------------------------------------------------
__global__ __launch_bounds__(32) void attn_fused(
    const __bf16* __restrict__ Q,   // [B, NH,  T, HD]
    const __bf16* __restrict__ Kc,  // [B, NKV, T, HD]
    const __bf16* __restrict__ VT,  // [B, NKV, HD, T]
    __bf16* __restrict__ O)         // [B, T, NH*HD]
{
  __shared__ __align__(16) __bf16 pbuf[16 * 32];

  const int lane = threadIdx.x;
  const int c  = lane & 15;
  const int hh = lane >> 4;
  const int qt   = blockIdx.x;      // 0..T/16-1
  const int head = blockIdx.y;      // 0..15
  const int b    = blockIdx.z;      // 0..3
  const int kvh  = head >> 2;       // KV_GROUPS = 4
  const float scale = 0.08838834764831845f;   // 1/sqrt(128)

  // Q tile as 4 A-operands (HD = 4 chunks of K=32)
  ABf16 qa[4];
  {
    const __bf16* qrow =
        Q + ((size_t)(b * NH + head) * Tsz + qt * 16 + c) * HDm;
#pragma unroll
    for (int cc = 0; cc < 4; ++cc) {
      qa[cc].h[0] = *(const bf16x8*)(qrow + 32 * cc + 16 * hh);
      qa[cc].h[1] = *(const bf16x8*)(qrow + 32 * cc + 16 * hh + 8);
    }
  }

  f32x8 o[8];
  float mrun[8], lrun[8];
#pragma unroll
  for (int nt = 0; nt < 8; ++nt) o[nt] = fzero8();
#pragma unroll
  for (int r = 0; r < 8; ++r) { mrun[r] = -__builtin_inff(); lrun[r] = 0.f; }

  const __bf16* kbase = Kc + (size_t)(b * NKV + kvh) * Tsz * HDm;
  const __bf16* vbase = VT + (size_t)(b * NKV + kvh) * HDm * Tsz;

  const int nblk = (16 * qt + 47) >> 5;   // KV blocks of 32 needed (causal)
  for (int j = 0; j < nblk; ++j) {
    const int kv0 = j * 32;

    // ---- S = Q * K^T (two 16x16 tiles) ----
    f32x8 s0 = fzero8(), s1 = fzero8();
    {
      const __bf16* kr0 = kbase + (size_t)(kv0 + c) * HDm;
      const __bf16* kr1 = kbase + (size_t)(kv0 + 16 + c) * HDm;
#pragma unroll
      for (int cc = 0; cc < 4; ++cc) {
        ABf16 kb0, kb1;
        kb0.h[0] = *(const bf16x8*)(kr0 + 32 * cc + 16 * hh);
        kb0.h[1] = *(const bf16x8*)(kr0 + 32 * cc + 16 * hh + 8);
        kb1.h[0] = *(const bf16x8*)(kr1 + 32 * cc + 16 * hh);
        kb1.h[1] = *(const bf16x8*)(kr1 + 32 * cc + 16 * hh + 8);
        s0 = wmma_bf16(qa[cc].v, kb0.v, s0);
        s1 = wmma_bf16(qa[cc].v, kb1.v, s1);
      }
    }

    // ---- causal mask + online softmax update ----
    float alpha[8];
#pragma unroll
    for (int r = 0; r < 8; ++r) {
      const int row  = qt * 16 + r + 8 * hh;
      const int col0 = kv0 + c;
      const int col1 = kv0 + 16 + c;
      float v0 = (col0 <= row) ? s0[r] * scale : -__builtin_inff();
      float v1 = (col1 <= row) ? s1[r] * scale : -__builtin_inff();
      float mx = fmaxf(v0, v1);
#pragma unroll
      for (int m = 1; m < 16; m <<= 1)
        mx = fmaxf(mx, __shfl_xor(mx, m, 32));
      const float nm = fmaxf(mrun[r], mx);
      const float al = __expf(mrun[r] - nm);
      v0 = __expf(v0 - nm);
      v1 = __expf(v1 - nm);
      float rs = v0 + v1;
#pragma unroll
      for (int m = 1; m < 16; m <<= 1)
        rs += __shfl_xor(rs, m, 32);
      lrun[r] = lrun[r] * al + rs;
      mrun[r] = nm;
      alpha[r] = al;
      s0[r] = v0;
      s1[r] = v1;
    }
#pragma unroll
    for (int nt = 0; nt < 8; ++nt)
#pragma unroll
      for (int r = 0; r < 8; ++r) o[nt][r] *= alpha[r];

    // ---- P: C-layout -> A-layout via LDS ----
    __syncthreads();
#pragma unroll
    for (int r = 0; r < 8; ++r) {
      pbuf[(r + 8 * hh) * 32 + c]      = (__bf16)s0[r];
      pbuf[(r + 8 * hh) * 32 + 16 + c] = (__bf16)s1[r];
    }
    __syncthreads();
    ABf16 pa;
    pa.h[0] = *(const bf16x8*)(&pbuf[c * 32 + 16 * hh]);
    pa.h[1] = *(const bf16x8*)(&pbuf[c * 32 + 16 * hh + 8]);

    // ---- O += P * V  (8 hd tiles of 16) ----
#pragma unroll
    for (int nt = 0; nt < 8; ++nt) {
      const __bf16* vr = vbase + (size_t)(nt * 16 + c) * Tsz + kv0 + 16 * hh;
      ABf16 vb;
      vb.h[0] = *(const bf16x8*)(vr);
      vb.h[1] = *(const bf16x8*)(vr + 8);
      o[nt] = wmma_bf16(pa.v, vb.v, o[nt]);
    }
  }

  // ---- normalize + store [b, t, head*128 + hd] as bf16 ----
#pragma unroll
  for (int r = 0; r < 8; ++r) lrun[r] = 1.0f / lrun[r];
#pragma unroll
  for (int nt = 0; nt < 8; ++nt)
#pragma unroll
    for (int r = 0; r < 8; ++r) {
      const int t = qt * 16 + r + 8 * hh;
      O[((size_t)(b * Tsz + t)) * Dsz + head * HDm + nt * 16 + c] =
          (__bf16)(o[nt][r] * lrun[r]);
    }
}

// ---------------------------------------------------------------------------
// Launch
// ---------------------------------------------------------------------------
extern "C" void kernel_launch(void* const* d_in, const int* in_sizes, int n_in,
                              void* d_out, int out_size, void* d_ws, size_t ws_size,
                              hipStream_t stream)
{
  const float* x  = (const float*)d_in[0];
  const float* Wq = (const float*)d_in[1];
  const float* Wk = (const float*)d_in[2];
  const float* Wv = (const float*)d_in[3];
  const float* Wo = (const float*)d_in[4];
  float* out = (float*)d_out;

  char* p = (char*)d_ws;
  auto take = [&](size_t bytes) -> char* {
    char* r = p;
    p += (bytes + 255) & ~(size_t)255;
    return r;
  };

  const size_t M = (size_t)Bsz * Tsz;                 // 8192 rows
  __bf16* xb  = (__bf16*)take(M * Dsz * 2);
  __bf16* Wqb = (__bf16*)take((size_t)Dsz * Dsz * 2);
  __bf16* Wkb = (__bf16*)take((size_t)(NKV * HDm) * Dsz * 2);
  __bf16* Wvb = (__bf16*)take((size_t)(NKV * HDm) * Dsz * 2);
  __bf16* Wob = (__bf16*)take((size_t)Dsz * Dsz * 2);
  float*  qf  = (float*) take(M * Dsz * 4);
  float*  kf  = (float*) take(M * (NKV * HDm) * 4);
  float*  vf  = (float*) take(M * (NKV * HDm) * 4);
  __bf16* qbf = (__bf16*)take(M * Dsz * 2);
  __bf16* kbf = (__bf16*)take(M * (NKV * HDm) * 2);
  __bf16* vTb = (__bf16*)take(M * (NKV * HDm) * 2);
  __bf16* aob = (__bf16*)take(M * Dsz * 2);

  auto cvt = [&](const float* src, __bf16* dst, int n) {
    cvt_f32_to_bf16<<<(n + 255) / 256, 256, 0, stream>>>(src, dst, n);
  };
  cvt(x,  xb,  (int)(M * Dsz));
  cvt(Wq, Wqb, Dsz * Dsz);
  cvt(Wk, Wkb, NKV * HDm * Dsz);
  cvt(Wv, Wvb, NKV * HDm * Dsz);
  cvt(Wo, Wob, Dsz * Dsz);

  // Projections (bf16 WMMA, f32 out)
  gemm_bf16_nt<<<dim3(Dsz / 32, (unsigned)(M / 256)), 128, 0, stream>>>(
      xb, Wqb, qf, (int)M, Dsz, Dsz);
  gemm_bf16_nt<<<dim3((NKV * HDm) / 32, (unsigned)(M / 256)), 128, 0, stream>>>(
      xb, Wkb, kf, (int)M, NKV * HDm, Dsz);
  gemm_bf16_nt<<<dim3((NKV * HDm) / 32, (unsigned)(M / 256)), 128, 0, stream>>>(
      xb, Wvb, vf, (int)M, NKV * HDm, Dsz);

  // RoPE + pack, V transpose-pack
  rope_pack<<<(Bsz * NH  * Tsz * 64) / 256, 256, 0, stream>>>(qf, qbf, NH);
  rope_pack<<<(Bsz * NKV * Tsz * 64) / 256, 256, 0, stream>>>(kf, kbf, NKV);
  v_pack_T <<<(Bsz * NKV * HDm * Tsz) / 256, 256, 0, stream>>>(vf, vTb);

  // Fused causal attention
  attn_fused<<<dim3(Tsz / 16, NH, Bsz), 32, 0, stream>>>(qbf, kbf, vTb, aob);

  // Output projection -> f32 d_out
  gemm_bf16_nt<<<dim3(Dsz / 32, (unsigned)(M / 256)), 128, 0, stream>>>(
      aob, Wob, out, (int)M, Dsz, Dsz);

  (void)in_sizes; (void)n_in; (void)out_size; (void)ws_size;
}
